// TrafficFoundationModel_82643760710111
// MI455X (gfx1250) — matrix operand order repaired and checked by reading, not compiled, not run on hardware
//
#include <hip/hip_runtime.h>

// ---------------------------------------------------------------------------
// Types for CDNA5 WMMA (wave32). bf16 handled as raw ushort bit patterns.
// ---------------------------------------------------------------------------
typedef __attribute__((ext_vector_type(16))) __bf16 v16bf;
typedef __attribute__((ext_vector_type(8)))  float  v8f;

union FragA { v16bf v; unsigned u[8]; };
union AccF  { v8f   v; float    f[8]; };

__device__ __forceinline__ unsigned short f2b(float f) {
  unsigned u = __float_as_uint(f);
  unsigned r = u + 0x7FFFu + ((u >> 16) & 1u);   // round-to-nearest-even
  return (unsigned short)(r >> 16);
}
__device__ __forceinline__ float b2f(unsigned short h) {
  return __uint_as_float(((unsigned)h) << 16);
}

constexpr int kBATCH = 4, kS = 1024, kD = 768, kH = 12, kL = 2, kE = 8,
              kF = 3072, kHD = 64;
constexpr long long kT = (long long)kBATCH * kS;   // 4096 tokens

// ---------------------------------------------------------------------------
// Generic WMMA GEMM: C[z] (+)= scale * rowscale[m] * (A[z] @ B[z]) + bias
//   A: bf16 [M x Kd], row stride lda   (M must be a multiple of 128)
//   B: TB = float (weights, converted on the fly) or ushort (bf16 acts)
//      BT=false: B is [Kd x N] row-major;  BT=true: B is [N x Kd] (i.e. B^T)
//   Batch z: base offsets (z/ZH)*outer + (z%ZH)*inner per operand.
//   Tile: 128x64 per 256-thread block (8 waves). Each wave owns a 2x2 grid of
//   16x16 WMMA accumulators -> 4 v_wmma per K-chunk sharing 2 A + 2 B frags.
// ---------------------------------------------------------------------------
template<typename TB, bool BT>
__global__ __launch_bounds__(256)
void gemm_wmma(const unsigned short* __restrict__ A, const TB* __restrict__ B,
               float* __restrict__ C, int M, int N, int Kd,
               int lda, int ldb, int ldc,
               long long sAo, long long sAi, long long sBo, long long sBi,
               long long sCo, long long sCi, int ZH,
               const float* __restrict__ bias,
               const float* __restrict__ rowscale, int rs_stride,
               float scale, int addC, int causal)
{
  (void)M; (void)N;
  __shared__ __align__(16) unsigned short lsA[128][40];  // [m][k], padded
  __shared__ __align__(16) unsigned short lsB[64][40];   // [n][k], padded
  int z = blockIdx.z, zo = z / ZH, zi = z % ZH;
  A += zo * sAo + zi * sAi;
  const TB* Bb = B + zo * sBo + zi * sBi;
  C += zo * sCo + zi * sCi;
  int m0 = blockIdx.y * 128, n0 = blockIdx.x * 64;
  if (causal && n0 > m0 + 127) return;                   // fully masked tile
  int tid = threadIdx.x, lane = tid & 31, wave = tid >> 5;
  int mtb = (wave & 3) * 2, ntb = (wave >> 2) * 2;       // wave 2x2 tile base
  int half = lane >> 4, lr = lane & 15;
  AccF acc[2][2];
  #pragma unroll
  for (int mi = 0; mi < 2; mi++)
    #pragma unroll
    for (int ni = 0; ni < 2; ni++)
      #pragma unroll
      for (int i = 0; i < 8; i++) acc[mi][ni].f[i] = 0.f;

  for (int k0 = 0; k0 < Kd; k0 += 32) {
    { // stage A tile 128x32 bf16: 2x 16B per thread
      int seg = tid & 3;
      #pragma unroll
      for (int rep = 0; rep < 2; rep++) {
        int r = (tid >> 2) + rep * 64;
        const unsigned short* src =
            A + (long long)(m0 + r) * lda + k0 + seg * 8;
        *(uint4*)&lsA[r][seg * 8] = *(const uint4*)src;
        if (k0 + 32 < Kd) __builtin_prefetch(src + 32, 0, 1);  // global_prefetch_b8
      }
    }
    if constexpr (BT) { // B stored [N x Kd] bf16 (e.g. K for Q@K^T)
      int r = tid >> 2, seg = tid & 3;
      const unsigned short* src =
          (const unsigned short*)Bb + (long long)(n0 + r) * ldb + k0 + seg * 8;
      *(uint4*)&lsB[r][seg * 8] = *(const uint4*)src;
    } else {            // B stored [Kd x N]; transpose into LDS, cvt to bf16
      int kk = tid >> 3, seg = tid & 7;
      const TB* src = Bb + (long long)(k0 + kk) * ldb + n0 + seg * 8;
      #pragma unroll
      for (int j = 0; j < 8; j++) {
        float f;
        if constexpr (sizeof(TB) == 2) f = b2f((unsigned short)src[j]);
        else                           f = (float)src[j];
        lsB[seg * 8 + j][kk] = f2b(f);
      }
    }
    __syncthreads();

    // Build fragments per ISA 7.12.2 VGPR layouts (wave32):
    //  A 16x32: lane holds row M=lr; VGPR i -> K = (i/4)*16 + half*8 + (i%4)*2
    //  B 32x16: lane holds col N=lr; VGPR i -> K = (i/4)*16 + half*8 + (i%4)*2
    FragA a0, a1, b0, b1;
    #pragma unroll
    for (int i = 0; i < 8; i++) {
      int ka = ((i >> 2) << 4) + (half << 3) + ((i & 3) << 1);
      a0.u[i] = *(const unsigned*)&lsA[mtb * 16 + lr][ka];
      a1.u[i] = *(const unsigned*)&lsA[mtb * 16 + 16 + lr][ka];
      int kb = ((i >> 2) << 4) + (half << 3) + ((i & 3) << 1);
      b0.u[i] = *(const unsigned*)&lsB[ntb * 16 + lr][kb];
      b1.u[i] = *(const unsigned*)&lsB[ntb * 16 + 16 + lr][kb];
    }
    acc[0][0].v = __builtin_amdgcn_wmma_f32_16x16x32_bf16(
        false, a0.v, false, b0.v, (short)0, acc[0][0].v, false, false);
    acc[0][1].v = __builtin_amdgcn_wmma_f32_16x16x32_bf16(
        false, a0.v, false, b1.v, (short)0, acc[0][1].v, false, false);
    acc[1][0].v = __builtin_amdgcn_wmma_f32_16x16x32_bf16(
        false, a1.v, false, b0.v, (short)0, acc[1][0].v, false, false);
    acc[1][1].v = __builtin_amdgcn_wmma_f32_16x16x32_bf16(
        false, a1.v, false, b1.v, (short)0, acc[1][1].v, false, false);
    __syncthreads();
  }

  // Epilogue: C/D layout -> VGPR i holds (M = i + half*8, N = lane&15)
  #pragma unroll
  for (int mi = 0; mi < 2; mi++) {
    #pragma unroll
    for (int i = 0; i < 8; i++) {
      int row = m0 + (mtb + mi) * 16 + i + half * 8;
      float rsv = rowscale ? rowscale[(long long)row * rs_stride] : 1.f;
      #pragma unroll
      for (int ni = 0; ni < 2; ni++) {
        int col = n0 + (ntb + ni) * 16 + lr;
        float r = acc[mi][ni].f[i] * scale * rsv;
        if (bias) r += bias[col];
        long long o = (long long)row * ldc + col;
        C[o] = (addC ? C[o] : 0.f) + r;
      }
    }
  }
}

// ---------------------------------------------------------------------------
// Embedding gather + concat -> bf16 [T, 3D]
// ---------------------------------------------------------------------------
__global__ void embed_concat(
    const int* ids, const int* ts, const int* dow, const int* mon,
    const int* hol, const int* loc, const int* road, const int* wst,
    const float* word_e, const float* time_e, const float* dow_e,
    const float* month_e, const float* hol_e, const float* loc_e,
    const float* road_e, const float* wth_e, unsigned short* out)
{
  long long i = (long long)blockIdx.x * 256 + threadIdx.x;
  if (i >= kT * (3 * kD)) return;
  int t = (int)(i / (3 * kD)), c = (int)(i % (3 * kD));
  int b = t / kS;
  const int Q = kD / 4;  // 192
  float v;
  if (c < kD) {
    v = word_e[(long long)ids[t] * kD + c];
  } else if (c < 2 * kD) {
    int c2 = c - kD, seg = c2 / Q, off = c2 % Q;
    if      (seg == 0) v = time_e[ts[b] * Q + off];
    else if (seg == 1) v = dow_e[dow[b] * Q + off];
    else if (seg == 2) v = month_e[mon[b] * Q + off];
    else               v = hol_e[hol[b] * Q + off];
  } else if (c < 2 * kD + 2 * Q) {
    v = loc_e[loc[b] * (2 * Q) + (c - 2 * kD)];
  } else {
    int c3 = c - 2 * kD - 2 * Q;
    v = (c3 < Q) ? road_e[road[b] * Q + c3] : wth_e[wst[b] * Q + (c3 - Q)];
  }
  out[i] = f2b(v);
}

// ---------------------------------------------------------------------------
// RMSNorm: one 256-thread block per row of D=768. Optional f32/bf16 outputs.
// ---------------------------------------------------------------------------
__global__ __launch_bounds__(256)
void rmsnorm_k(const float* x, const float* w, float* outf, unsigned short* outb)
{
  int row = blockIdx.x;
  const float* xr = x + (long long)row * kD;
  __shared__ float red[256];
  float s = 0.f;
  for (int c = threadIdx.x; c < kD; c += 256) { float v = xr[c]; s += v * v; }
  red[threadIdx.x] = s; __syncthreads();
  for (int st = 128; st > 0; st >>= 1) {
    if (threadIdx.x < st) red[threadIdx.x] += red[threadIdx.x + st];
    __syncthreads();
  }
  float inv = rsqrtf(red[0] / (float)kD + 1e-6f);
  for (int c = threadIdx.x; c < kD; c += 256) {
    float v = xr[c] * inv * w[c];
    if (outf) outf[(long long)row * kD + c] = v;
    if (outb) outb[(long long)row * kD + c] = f2b(v);
  }
}

// ---------------------------------------------------------------------------
// RoPE on q,k + bf16 convert of q,k,v. Layout [b,s,h,hd] flattened [T,D].
// ---------------------------------------------------------------------------
__global__ void rope_convert(const float* q, const float* k, const float* v,
                             unsigned short* qb, unsigned short* kb,
                             unsigned short* vb)
{
  long long i = (long long)blockIdx.x * 256 + threadIdx.x;
  if (i >= kT * kD) return;
  int t = (int)(i / kD), d = (int)(i % kD);
  int spos = t % kS;
  int hd = d & (kHD - 1);
  int j = hd & 31;
  // inv_freq[j] = 10000^(-2j/64) = exp(-ln(1e4)/32 * j)
  float f = (float)spos * __expf(-0.28782313662f * (float)j);
  float cs = __cosf(f), sn = __sinf(f);
  float qv = q[i], kv = k[i], qr, kr;
  if (hd < 32) { qr = -q[i + 32]; kr = -k[i + 32]; }
  else         { qr =  q[i - 32]; kr =  k[i - 32]; }
  qb[i] = f2b(qv * cs + qr * sn);
  kb[i] = f2b(kv * cs + kr * sn);
  vb[i] = f2b(v[i]);
}

// ---------------------------------------------------------------------------
// Causal softmax over scores [BH*S, S] -> bf16 probs (masked cols = 0)
// ---------------------------------------------------------------------------
__global__ __launch_bounds__(256)
void softmax_causal(const float* sc, unsigned short* pr)
{
  int row = blockIdx.x;
  int srow = row % kS;
  const float* r = sc + (long long)row * kS;
  unsigned short* o = pr + (long long)row * kS;
  __shared__ float red[256];
  float mx = -1e30f;
  for (int c = threadIdx.x; c <= srow; c += 256) mx = fmaxf(mx, r[c]);
  red[threadIdx.x] = mx; __syncthreads();
  for (int st = 128; st > 0; st >>= 1) {
    if (threadIdx.x < st) red[threadIdx.x] = fmaxf(red[threadIdx.x], red[threadIdx.x + st]);
    __syncthreads();
  }
  mx = red[0]; __syncthreads();
  float sm = 0.f;
  for (int c = threadIdx.x; c <= srow; c += 256) sm += __expf(r[c] - mx);
  red[threadIdx.x] = sm; __syncthreads();
  for (int st = 128; st > 0; st >>= 1) {
    if (threadIdx.x < st) red[threadIdx.x] += red[threadIdx.x + st];
    __syncthreads();
  }
  float inv = 1.f / red[0];
  for (int c = threadIdx.x; c < kS; c += 256)
    o[c] = (c <= srow) ? f2b(__expf(r[c] - mx) * inv) : (unsigned short)0;
}

// ---------------------------------------------------------------------------
// Elementwise helpers
// ---------------------------------------------------------------------------
__global__ void f2bf_k(const float* in, unsigned short* out, long long n) {
  long long i = (long long)blockIdx.x * 256 + threadIdx.x;
  if (i < n) out[i] = f2b(in[i]);
}
__global__ void silumul_k(const float* g, const float* u, unsigned short* out,
                          long long n) {
  long long i = (long long)blockIdx.x * 256 + threadIdx.x;
  if (i < n) {
    float gv = g[i];
    out[i] = f2b(gv / (1.f + __expf(-gv)) * u[i]);
  }
}
__global__ void zero_f(float* p, int n) {
  int i = blockIdx.x * 256 + threadIdx.x;
  if (i < n) p[i] = 0.f;
}
__global__ void copy1(const float* src, float* dst) { dst[0] = src[0]; }

// ---------------------------------------------------------------------------
// Router: one wave32 per token. Softmax over E=8, top-2 renormalized combine,
// atomic aux stats: stats[0..7]=sum probs, stats[8..15]=top-k counts.
// ---------------------------------------------------------------------------
__global__ __launch_bounds__(256)
void router_k(const unsigned short* xn, const float* rw, float* combine,
              float* stats)
{
  int lane = threadIdx.x & 31;
  int row = blockIdx.x * 8 + (threadIdx.x >> 5);
  float acc[kE];
  #pragma unroll
  for (int e = 0; e < kE; e++) acc[e] = 0.f;
  for (int c = lane; c < kD; c += 32) {
    float xv = b2f(xn[(long long)row * kD + c]);
    #pragma unroll
    for (int e = 0; e < kE; e++) acc[e] += xv * rw[c * kE + e];
  }
  #pragma unroll
  for (int off = 16; off > 0; off >>= 1)
    #pragma unroll
    for (int e = 0; e < kE; e++) acc[e] += __shfl_down(acc[e], off, 32);
  if (lane == 0) {
    float mx = acc[0];
    #pragma unroll
    for (int e = 1; e < kE; e++) mx = fmaxf(mx, acc[e]);
    float p[kE], sum = 0.f;
    #pragma unroll
    for (int e = 0; e < kE; e++) { p[e] = __expf(acc[e] - mx); sum += p[e]; }
    #pragma unroll
    for (int e = 0; e < kE; e++) p[e] /= sum;
    int i1 = 0;
    #pragma unroll
    for (int e = 1; e < kE; e++) if (p[e] > p[i1]) i1 = e;
    int i2 = (i1 == 0) ? 1 : 0;
    #pragma unroll
    for (int e = 0; e < kE; e++) if (e != i1 && p[e] > p[i2]) i2 = e;
    float wsum = p[i1] + p[i2];
    #pragma unroll
    for (int e = 0; e < kE; e++)
      combine[(long long)row * kE + e] =
          (e == i1) ? p[i1] / wsum : ((e == i2) ? p[i2] / wsum : 0.f);
    #pragma unroll
    for (int e = 0; e < kE; e++) atomicAdd(&stats[e], p[e]);
    atomicAdd(&stats[kE + i1], 1.f);
    atomicAdd(&stats[kE + i2], 1.f);
  }
}

__global__ void aux_fin(const float* stats, float* aux) {
  if (threadIdx.x == 0 && blockIdx.x == 0) {
    float s = 0.f;
    for (int e = 0; e < kE; e++)
      s += (stats[kE + e] / (float)kT) * (stats[e] / (float)kT);
    aux[0] += s * (float)kE;
  }
}

// ---------------------------------------------------------------------------
// Host: orchestrate the whole forward on `stream`.
// ---------------------------------------------------------------------------
extern "C" void kernel_launch(void* const* d_in, const int* in_sizes, int n_in,
                              void* d_out, int out_size, void* d_ws,
                              size_t ws_size, hipStream_t stream)
{
  (void)in_sizes; (void)n_in; (void)out_size; (void)ws_size;
  // inputs per setup_inputs() order
  const int* ids  = (const int*)d_in[0];
  const int* ts   = (const int*)d_in[1];
  const int* dow  = (const int*)d_in[2];
  const int* mon  = (const int*)d_in[3];
  const int* hol  = (const int*)d_in[4];
  const int* loc  = (const int*)d_in[5];
  const int* road = (const int*)d_in[6];
  const int* wst  = (const int*)d_in[7];
  const float* word_e  = (const float*)d_in[8];
  const float* time_e  = (const float*)d_in[9];
  const float* dow_e   = (const float*)d_in[10];
  const float* month_e = (const float*)d_in[11];
  const float* hol_e   = (const float*)d_in[12];
  const float* loc_e   = (const float*)d_in[13];
  const float* road_e  = (const float*)d_in[14];
  const float* wth_e   = (const float*)d_in[15];
  const float* proj_w  = (const float*)d_in[16];
  const float* proj_b  = (const float*)d_in[17];
  const float* embn_w  = (const float*)d_in[18];
  const float* norm1_w = (const float*)d_in[19];
  const float* Wq_all  = (const float*)d_in[20];
  const float* Wk_all  = (const float*)d_in[21];
  const float* Wv_all  = (const float*)d_in[22];
  const float* Wo_all  = (const float*)d_in[23];
  const float* norm2_w = (const float*)d_in[24];
  const float* rtr_all = (const float*)d_in[25];
  const float* Wg_all  = (const float*)d_in[26];
  const float* Wu_all  = (const float*)d_in[27];
  const float* Wd_all  = (const float*)d_in[28];
  const float* finn_w  = (const float*)d_in[29];
  float* out = (float*)d_out;

  // workspace bump allocator
  unsigned char* wsb = (unsigned char*)d_ws;
  size_t off = 0;
  auto take = [&](size_t bytes) -> void* {
    void* p = wsb + off;
    off += (bytes + 255) & ~(size_t)255;
    return p;
  };
  float*          x_f    = (float*)take(kT * kD * 4);
  unsigned short* xn_b   = (unsigned short*)take(kT * kD * 2);
  unsigned short* cat_b  = (unsigned short*)take(kT * 3 * kD * 2);
  float*          q_f    = (float*)take(kT * kD * 4);
  float*          k_f    = (float*)take(kT * kD * 4);
  float*          v_f    = (float*)take(kT * kD * 4);
  unsigned short* q_b    = (unsigned short*)take(kT * kD * 2);
  unsigned short* k_b    = (unsigned short*)take(kT * kD * 2);
  unsigned short* v_b    = (unsigned short*)take(kT * kD * 2);
  float*          sc_f   = (float*)take((size_t)kBATCH * kH * kS * kS * 4);
  unsigned short* pr_b   = (unsigned short*)take((size_t)kBATCH * kH * kS * kS * 2);
  float*          attn_f = (float*)take(kT * kD * 4);
  unsigned short* attn_b = (unsigned short*)take(kT * kD * 2);
  float*          g_f    = (float*)take(kT * kF * 4);
  float*          u_f    = (float*)take(kT * kF * 4);
  unsigned short* h_b    = (unsigned short*)take(kT * kF * 2);
  float*          comb   = (float*)take(kT * kE * 4);
  float*          stats  = (float*)take(64);
  float*          aux    = (float*)take(16);

  auto blks = [](long long n) { return (unsigned)((n + 255) / 256); };
  const long long SS = (long long)kS * kS;

  zero_f<<<1, 256, 0, stream>>>(aux, 1);

  // ---- embedding: concat -> proj (+bias) -> rmsnorm ----
  embed_concat<<<blks(kT * 3 * kD), 256, 0, stream>>>(
      ids, ts, dow, mon, hol, loc, road, wst, word_e, time_e, dow_e, month_e,
      hol_e, loc_e, road_e, wth_e, cat_b);
  gemm_wmma<float, false><<<dim3(kD / 64, (unsigned)(kT / 128), 1), 256, 0, stream>>>(
      cat_b, proj_w, x_f, (int)kT, kD, 3 * kD, 3 * kD, kD, kD,
      0, 0, 0, 0, 0, 0, 1, proj_b, nullptr, 0, 1.f, 0, 0);
  rmsnorm_k<<<(unsigned)kT, 256, 0, stream>>>(x_f, embn_w, x_f, nullptr);

  for (int l = 0; l < kL; l++) {
    const float* Wq = Wq_all + (long long)l * kD * kD;
    const float* Wk = Wk_all + (long long)l * kD * kD;
    const float* Wv = Wv_all + (long long)l * kD * kD;
    const float* Wo = Wo_all + (long long)l * kD * kD;

    // x -> rmsnorm -> Q,K,V
    rmsnorm_k<<<(unsigned)kT, 256, 0, stream>>>(x_f, norm1_w + l * kD, nullptr, xn_b);
    gemm_wmma<float, false><<<dim3(kD / 64, (unsigned)(kT / 128), 1), 256, 0, stream>>>(
        xn_b, Wq, q_f, (int)kT, kD, kD, kD, kD, kD,
        0, 0, 0, 0, 0, 0, 1, nullptr, nullptr, 0, 1.f, 0, 0);
    gemm_wmma<float, false><<<dim3(kD / 64, (unsigned)(kT / 128), 1), 256, 0, stream>>>(
        xn_b, Wk, k_f, (int)kT, kD, kD, kD, kD, kD,
        0, 0, 0, 0, 0, 0, 1, nullptr, nullptr, 0, 1.f, 0, 0);
    gemm_wmma<float, false><<<dim3(kD / 64, (unsigned)(kT / 128), 1), 256, 0, stream>>>(
        xn_b, Wv, v_f, (int)kT, kD, kD, kD, kD, kD,
        0, 0, 0, 0, 0, 0, 1, nullptr, nullptr, 0, 1.f, 0, 0);
    rope_convert<<<blks(kT * kD), 256, 0, stream>>>(q_f, k_f, v_f, q_b, k_b, v_b);

    // scores = (Q @ K^T) / 8, batched over (b,h), causal tiles skipped
    gemm_wmma<unsigned short, true><<<dim3(kS / 64, kS / 128, kBATCH * kH), 256, 0, stream>>>(
        q_b, k_b, sc_f, kS, kS, kHD, kD, kD, kS,
        (long long)kS * kD, kHD, (long long)kS * kD, kHD,
        (long long)kH * SS, SS, kH, nullptr, nullptr, 0, 0.125f, 0, 1);
    softmax_causal<<<(unsigned)(kBATCH * kH * kS), 256, 0, stream>>>(sc_f, pr_b);

    // attn = P @ V  -> [b,s,h,hd] layout
    gemm_wmma<unsigned short, false><<<dim3(1, kS / 128, kBATCH * kH), 256, 0, stream>>>(
        pr_b, v_b, attn_f, kS, kHD, kS, kS, kD, kD,
        (long long)kH * SS, SS, (long long)kS * kD, kHD,
        (long long)kS * kD, kHD, kH, nullptr, nullptr, 0, 1.f, 0, 0);
    f2bf_k<<<blks(kT * kD), 256, 0, stream>>>(attn_f, attn_b, kT * kD);

    // x += attn @ Wo
    gemm_wmma<float, false><<<dim3(kD / 64, (unsigned)(kT / 128), 1), 256, 0, stream>>>(
        attn_b, Wo, x_f, (int)kT, kD, kD, kD, kD, kD,
        0, 0, 0, 0, 0, 0, 1, nullptr, nullptr, 0, 1.f, 1, 0);

    // ---- MoE ----
    rmsnorm_k<<<(unsigned)kT, 256, 0, stream>>>(x_f, norm2_w + l * kD, nullptr, xn_b);
    zero_f<<<1, 256, 0, stream>>>(stats, 16);
    router_k<<<(unsigned)(kT / 8), 256, 0, stream>>>(
        xn_b, rtr_all + (long long)l * kD * kE, comb, stats);
    aux_fin<<<1, 32, 0, stream>>>(stats, aux);

    for (int e = 0; e < kE; e++) {
      const float* Wg = Wg_all + ((long long)l * kE + e) * kD * kF;
      const float* Wu = Wu_all + ((long long)l * kE + e) * kD * kF;
      const float* Wd = Wd_all + ((long long)l * kE + e) * kF * kD;
      gemm_wmma<float, false><<<dim3(kF / 64, (unsigned)(kT / 128), 1), 256, 0, stream>>>(
          xn_b, Wg, g_f, (int)kT, kF, kD, kD, kF, kF,
          0, 0, 0, 0, 0, 0, 1, nullptr, nullptr, 0, 1.f, 0, 0);
      gemm_wmma<float, false><<<dim3(kF / 64, (unsigned)(kT / 128), 1), 256, 0, stream>>>(
          xn_b, Wu, u_f, (int)kT, kF, kD, kD, kF, kF,
          0, 0, 0, 0, 0, 0, 1, nullptr, nullptr, 0, 1.f, 0, 0);
      silumul_k<<<blks(kT * kF), 256, 0, stream>>>(g_f, u_f, h_b, kT * kF);
      // x += combine[:,e] * (h @ Wd)
      gemm_wmma<float, false><<<dim3(kD / 64, (unsigned)(kT / 128), 1), 256, 0, stream>>>(
          h_b, Wd, x_f, (int)kT, kD, kF, kF, kD, kD,
          0, 0, 0, 0, 0, 0, 1, nullptr, comb + e, kE, 1.f, 1, 0);
    }
  }

  // final rmsnorm -> d_out[0 .. T*D), aux -> d_out[T*D]
  rmsnorm_k<<<(unsigned)kT, 256, 0, stream>>>(x_f, finn_w, out, nullptr);
  copy1<<<1, 1, 0, stream>>>(aux, out + kT * kD);
}
